// PJAC_65214783422851
// MI455X (gfx1250) — matrix operand order repaired and checked by reading, not compiled, not run on hardware
//
#include <hip/hip_runtime.h>
#include <hip/hip_bf16.h>

typedef __attribute__((ext_vector_type(16))) __bf16 v16bf;
typedef __attribute__((ext_vector_type(8)))  float  v8f;

#define DIMX 1024
#define NH   16
#define HDIM 64
#define BB   2
#define PP   128
#define TTOK 2048
#define SSEQ 2176   // PP + TTOK
#define C3   3072   // 3*NH*HDIM

// ---------- helpers ----------

__device__ __forceinline__ unsigned short f2bf(float f) {
    unsigned int u = __float_as_uint(f);
    u += 0x7FFFu + ((u >> 16) & 1u);          // round-to-nearest-even
    return (unsigned short)(u >> 16);
}

union BFrag { v16bf v; uint4 q[2]; unsigned short us[16]; };

// 16 contiguous bf16 (B-fragment per-lane data)
__device__ __forceinline__ v16bf load_bfrag_contig(const unsigned short* p) {
    BFrag f;
    f.q[0] = *(const uint4*)(p);
    f.q[1] = *(const uint4*)(p + 8);
    return f.v;
}
// A-fragment per-lane data: 16 contiguous bf16 at p (contiguous K order, matches B)
__device__ __forceinline__ v16bf load_afrag_split(const unsigned short* p) {
    BFrag f;
    f.q[0] = *(const uint4*)(p);
    f.q[1] = *(const uint4*)(p + 8);
    return f.v;
}

#define WMMA_BF16(A, B, C) \
    __builtin_amdgcn_wmma_f32_16x16x32_bf16(false, (A), false, (B), (short)0, (C), false, false)

// ---------- kernel 1: weight transpose + f32->bf16 ----------
// W is K x N row-major f32; Wt is N x K row-major bf16
__global__ void k_transpose_bf16(const float* __restrict__ W,
                                 unsigned short* __restrict__ Wt,
                                 int K, int N) {
    int idx = blockIdx.x * blockDim.x + threadIdx.x;
    if (idx >= K * N) return;
    int k = idx / N, n = idx - k * N;
    Wt[(size_t)n * K + k] = f2bf(W[idx]);
}

// ---------- kernel 2: row RMSNorm -> bf16 ----------
__global__ void k_rmsnorm_bf16(const float* __restrict__ X,
                               const float* __restrict__ w,
                               unsigned short* __restrict__ A) {
    int row = blockIdx.x;
    int tid = threadIdx.x;                     // 256 threads, 4 elems each
    const float* xr = X + (size_t)row * DIMX;
    float4 v = *(const float4*)(xr + tid * 4);
    float ss = v.x * v.x + v.y * v.y + v.z * v.z + v.w * v.w;
    for (int m = 16; m >= 1; m >>= 1) ss += __shfl_xor(ss, m, 32);
    __shared__ float red[8];
    if ((tid & 31) == 0) red[tid >> 5] = ss;
    __syncthreads();
    float tot = 0.f;
#pragma unroll
    for (int i = 0; i < 8; i++) tot += red[i];
    float sc = rsqrtf(tot / (float)DIMX + 1e-6f);
    float4 wv = *(const float4*)(w + tid * 4);
    unsigned short* ar = A + (size_t)row * DIMX + tid * 4;
    ar[0] = f2bf(v.x * sc * wv.x);
    ar[1] = f2bf(v.y * sc * wv.y);
    ar[2] = f2bf(v.z * sc * wv.z);
    ar[3] = f2bf(v.w * sc * wv.w);
}

// ---------- kernel 3: QKV WMMA GEMM + bias + head-RMSNorm + RoPE ----------
// One wave per (16-row tile) x (head slot). slot: 0..15 -> Q heads, 16..31 -> K heads, 32..47 -> V heads
__global__ void __launch_bounds__(256)
k_qkv_wmma(const unsigned short* __restrict__ A,    // rows_total x 1024 bf16
           const unsigned short* __restrict__ Wt,   // 3072 x 1024 bf16
           const float* __restrict__ bias,          // 3072
           const float* __restrict__ qnw,           // 64
           const float* __restrict__ knw,           // 64
           unsigned short* __restrict__ Qb,         // (B,H,S,HD) bf16
           unsigned short* __restrict__ Kb,         // (B,H,S,HD) bf16
           unsigned short* __restrict__ Vt,         // (B,H,HD,S) bf16
           int rowsPerBatch, int sOff, int nWaveTotal) {
    int wid  = threadIdx.x >> 5;
    int lane = threadIdx.x & 31;
    int wg   = blockIdx.x * 8 + wid;
    if (wg >= nWaveTotal) return;

    int rowtile = wg / 48;
    int slot    = wg - rowtile * 48;
    int third   = slot >> 4;                  // 0=q 1=k 2=v
    int h       = slot & 15;
    int c0      = third * DIMX + h * HDIM;

    int grow0 = rowtile * 16;
    int b     = grow0 / rowsPerBatch;
    int r0    = grow0 - b * rowsPerBatch;

    int l15  = lane & 15;
    int hf   = lane >> 4;
    int klo  = hf * 8;
    int kb16 = hf * 16;
    int half8 = hf * 8;

    const unsigned short* arow = A + (size_t)(grow0 + l15) * DIMX + kb16;
    const unsigned short* wcol[4];
#pragma unroll
    for (int t = 0; t < 4; t++)
        wcol[t] = Wt + (size_t)(c0 + t * 16 + l15) * DIMX + kb16;

    v8f acc[4];
#pragma unroll
    for (int t = 0; t < 4; t++)
#pragma unroll
        for (int i = 0; i < 8; i++) acc[t][i] = 0.f;

    for (int kk = 0; kk < DIMX; kk += 32) {
        v16bf a = load_afrag_split(arow + kk);
#pragma unroll
        for (int t = 0; t < 4; t++) {
            v16bf bm = load_bfrag_contig(wcol[t] + kk);
            acc[t] = WMMA_BF16(a, bm, acc[t]);
        }
    }

    // bias
#pragma unroll
    for (int t = 0; t < 4; t++) {
        float bv = bias[c0 + t * 16 + l15];
#pragma unroll
        for (int i = 0; i < 8; i++) acc[t][i] += bv;
    }

    size_t bh = (size_t)(b * NH + h);

    if (third == 2) {
        // V: store transposed (HD x S); the 8 row-values per lane are contiguous
        // in S, so pack into one 16-byte store (global_store_b128).
#pragma unroll
        for (int t = 0; t < 4; t++) {
            int n = t * 16 + l15;
            unsigned short* vp = Vt + (bh * HDIM + n) * SSEQ + sOff + r0 + half8;
            unsigned int p0 = (unsigned)f2bf(acc[t][0]) | ((unsigned)f2bf(acc[t][1]) << 16);
            unsigned int p1 = (unsigned)f2bf(acc[t][2]) | ((unsigned)f2bf(acc[t][3]) << 16);
            unsigned int p2 = (unsigned)f2bf(acc[t][4]) | ((unsigned)f2bf(acc[t][5]) << 16);
            unsigned int p3 = (unsigned)f2bf(acc[t][6]) | ((unsigned)f2bf(acc[t][7]) << 16);
            uint4 pk; pk.x = p0; pk.y = p1; pk.z = p2; pk.w = p3;
            *(uint4*)vp = pk;
        }
    } else {
        const float* nw = (third == 0) ? qnw : knw;
        float wv[4];
#pragma unroll
        for (int t = 0; t < 4; t++) wv[t] = nw[t * 16 + l15];

        // head RMSNorm: row sum-of-squares across the 16 lanes of each half
        float rinv[8];
#pragma unroll
        for (int i = 0; i < 8; i++) {
            float s = 0.f;
#pragma unroll
            for (int t = 0; t < 4; t++) s += acc[t][i] * acc[t][i];
#pragma unroll
            for (int m = 1; m < 16; m <<= 1) s += __shfl_xor(s, m, 32);
            rinv[i] = rsqrtf(s / (float)HDIM + 1e-6f);
        }

        unsigned short* Ob = (third == 0) ? Qb : Kb;
#pragma unroll
        for (int t = 0; t < 4; t++) {
            int n = t * 16 + l15;
            int j = n >> 1;
            // inv_freq = 10000^(-2j/64) = exp(-j * ln(10000)/32)
            float invf = __expf(-0.28782313662425572f * (float)j);
            bool neven = ((n & 1) == 0);
#pragma unroll
            for (int i = 0; i < 8; i++) {
                float v = acc[t][i] * rinv[i] * wv[t];
                float other = __shfl_xor(v, 1, 32);     // pair partner (2j <-> 2j+1)
                int pos = r0 + half8 + i;
                float ang = (float)pos * invf, sn, cs;
                __sincosf(ang, &sn, &cs);
                float res = neven ? (v * cs - other * sn) : (other * sn + v * cs);
                int s = sOff + pos;
                Ob[(bh * SSEQ + s) * HDIM + n] = f2bf(res);
            }
        }
    }
}

// ---------- kernel 4: flash attention with WMMA ----------
// One wave per (b, h, 16-row query tile). Online softmax over 32-key chunks.
__global__ void __launch_bounds__(256)
k_attn_wmma(const unsigned short* __restrict__ Qb,   // (B,H,S,HD)
            const unsigned short* __restrict__ Kb,   // (B,H,S,HD)
            const unsigned short* __restrict__ Vt,   // (B,H,HD,S)
            const unsigned char* __restrict__ mask,  // (B,1,S,S)
            unsigned short* __restrict__ attnA,      // (B,S,H*HD) bf16
            int nWaveTotal) {
    __shared__ alignas(16) unsigned short pbuf[8][16 * 40];  // stride 40 -> 16B aligned, conflict-free
    int wid  = threadIdx.x >> 5;
    int lane = threadIdx.x & 31;
    int wg   = blockIdx.x * 8 + wid;
    if (wg >= nWaveTotal) return;

    const int NQT = SSEQ / 16;                // 136
    int b   = wg / (NH * NQT);
    int rem = wg - b * NH * NQT;
    int h   = rem / NQT;
    int qt  = rem - h * NQT;

    int l15  = lane & 15;
    int hf   = lane >> 4;
    int klo  = hf * 8;
    int kb16 = hf * 16;
    int half8 = hf * 8;

    size_t bh = (size_t)(b * NH + h);
    const unsigned short* qrow = Qb + (bh * SSEQ + qt * 16 + l15) * HDIM + kb16;
    v16bf qa0 = load_afrag_split(qrow);        // head dims 0..31
    v16bf qa1 = load_afrag_split(qrow + 32);   // head dims 32..63

    v8f acc[4];
    float mrow[8], lrow[8];
#pragma unroll
    for (int i = 0; i < 8; i++) { mrow[i] = -3.4028235e38f; lrow[i] = 0.f; }
#pragma unroll
    for (int t = 0; t < 4; t++)
#pragma unroll
        for (int i = 0; i < 8; i++) acc[t][i] = 0.f;

    unsigned short* pw = &pbuf[wid][0];
    const float SCALE = 0.125f;                // 1/sqrt(64)

    for (int kc = 0; kc < SSEQ / 32; kc++) {
        int key0 = kc * 32;

        // prefetch next chunk's K rows and V rows (global_prefetch_b8)
        if (kc + 1 < SSEQ / 32) {
            const unsigned short* kn =
                Kb + (bh * SSEQ + key0 + 32 + lane) * HDIM;
            __builtin_prefetch(kn, 0, 3);
            const unsigned short* vn =
                Vt + (bh * HDIM + (lane & 31)) * SSEQ + key0 + 32;
            __builtin_prefetch(vn, 0, 3);
        }

        v8f sc[2];
#pragma unroll
        for (int sub = 0; sub < 2; sub++) {
            const unsigned short* krow =
                Kb + (bh * SSEQ + key0 + sub * 16 + l15) * HDIM + kb16;
            v16bf bk0 = load_bfrag_contig(krow);        // dims 0..31 slice
            v16bf bk1 = load_bfrag_contig(krow + 32);   // dims 32..63 slice
            v8f s;
#pragma unroll
            for (int i = 0; i < 8; i++) s[i] = 0.f;
            s = WMMA_BF16(qa0, bk0, s);
            s = WMMA_BF16(qa1, bk1, s);
            sc[sub] = s;
        }
        // scale + mask
#pragma unroll
        for (int sub = 0; sub < 2; sub++) {
            int key = key0 + sub * 16 + l15;
#pragma unroll
            for (int i = 0; i < 8; i++) {
                int sq = qt * 16 + half8 + i;
                float v = sc[sub][i] * SCALE;
                unsigned char mb = mask[((size_t)b * SSEQ + sq) * SSEQ + key];
                sc[sub][i] = mb ? v : -3.4028235e38f;
            }
        }
        // online softmax update
#pragma unroll
        for (int i = 0; i < 8; i++) {
            float mx = fmaxf(sc[0][i], sc[1][i]);
#pragma unroll
            for (int x = 1; x < 16; x <<= 1) mx = fmaxf(mx, __shfl_xor(mx, x, 32));
            float mnew = fmaxf(mrow[i], mx);
            float resc = __expf(mrow[i] - mnew);
            float p0 = __expf(sc[0][i] - mnew);
            float p1 = __expf(sc[1][i] - mnew);
            float ps = p0 + p1;
#pragma unroll
            for (int x = 1; x < 16; x <<= 1) ps += __shfl_xor(ps, x, 32);
            lrow[i] = lrow[i] * resc + ps;
            mrow[i] = mnew;
#pragma unroll
            for (int t = 0; t < 4; t++) acc[t][i] *= resc;
            int row = half8 + i;
            pw[row * 40 + l15]      = f2bf(p0);
            pw[row * 40 + 16 + l15] = f2bf(p1);
        }
        asm volatile("s_wait_dscnt 0" ::: "memory");
        // P (16x32) as A-fragment from LDS
        v16bf pa = load_afrag_split(pw + l15 * 40 + kb16);
        // P @ V
#pragma unroll
        for (int t = 0; t < 4; t++) {
            const unsigned short* vp =
                Vt + (bh * HDIM + t * 16 + l15) * SSEQ + key0 + kb16;
            v16bf bv = load_bfrag_contig(vp);
            acc[t] = WMMA_BF16(pa, bv, acc[t]);
        }
    }

    // finalize + store in (B,S,H*HD) layout for the out-proj GEMM
#pragma unroll
    for (int i = 0; i < 8; i++) {
        float inv = 1.0f / lrow[i];
#pragma unroll
        for (int t = 0; t < 4; t++) acc[t][i] *= inv;
    }
#pragma unroll
    for (int t = 0; t < 4; t++) {
        int n = t * 16 + l15;
#pragma unroll
        for (int i = 0; i < 8; i++) {
            int sq = qt * 16 + half8 + i;
            attnA[((size_t)b * SSEQ + sq) * DIMX + h * HDIM + n] = f2bf(acc[t][i]);
        }
    }
}

// ---------- kernel 5: output projection WMMA GEMM + bias + gate ----------
__global__ void __launch_bounds__(256)
k_outproj_wmma(const unsigned short* __restrict__ attnA,  // (B,S,1024) bf16
               const unsigned short* __restrict__ Wt,     // 1024 x 1024 bf16
               const float* __restrict__ bias,            // 1024
               const unsigned char* __restrict__ gate,    // (B, rowsPerBatch)
               float* __restrict__ out,                   // (B, rowsPerBatch, 1024)
               int rowsPerBatch, int sOff, int nWaveTotal) {
    int wid  = threadIdx.x >> 5;
    int lane = threadIdx.x & 31;
    int wg   = blockIdx.x * 8 + wid;
    if (wg >= nWaveTotal) return;

    int rowtile = wg >> 4;
    int slot    = wg & 15;
    int c0      = slot * 64;

    int grow0 = rowtile * 16;
    int b     = grow0 / rowsPerBatch;
    int r0    = grow0 - b * rowsPerBatch;

    int l15  = lane & 15;
    int hf   = lane >> 4;
    int klo  = hf * 8;
    int kb16 = hf * 16;
    int half8 = hf * 8;

    const unsigned short* arow =
        attnA + ((size_t)b * SSEQ + sOff + r0 + l15) * DIMX + kb16;
    const unsigned short* wcol[4];
#pragma unroll
    for (int t = 0; t < 4; t++)
        wcol[t] = Wt + (size_t)(c0 + t * 16 + l15) * DIMX + kb16;

    v8f acc[4];
#pragma unroll
    for (int t = 0; t < 4; t++)
#pragma unroll
        for (int i = 0; i < 8; i++) acc[t][i] = 0.f;

    for (int kk = 0; kk < DIMX; kk += 32) {
        v16bf a = load_afrag_split(arow + kk);
#pragma unroll
        for (int t = 0; t < 4; t++) {
            v16bf bm = load_bfrag_contig(wcol[t] + kk);
            acc[t] = WMMA_BF16(a, bm, acc[t]);
        }
    }

    float bv[4];
#pragma unroll
    for (int t = 0; t < 4; t++) bv[t] = bias[c0 + t * 16 + l15];

#pragma unroll
    for (int i = 0; i < 8; i++) {
        int r = r0 + half8 + i;
        float g = gate[(size_t)b * rowsPerBatch + r] ? 1.0f : 0.0f;
        float* op = out + ((size_t)b * rowsPerBatch + r) * DIMX + c0 + l15;
#pragma unroll
        for (int t = 0; t < 4; t++) op[t * 16] = (acc[t][i] + bv[t]) * g;
    }
}

// ---------- host ----------
extern "C" void kernel_launch(void* const* d_in, const int* in_sizes, int n_in,
                              void* d_out, int out_size, void* d_ws, size_t ws_size,
                              hipStream_t stream) {
    const float* pool        = (const float*)d_in[0];
    const float* x           = (const float*)d_in[1];
    // d_in[2] = regions (unused by reference)
    const unsigned char* t_mask    = (const unsigned char*)d_in[3];
    const unsigned char* n_mask    = (const unsigned char*)d_in[4];
    const unsigned char* attn_mask = (const unsigned char*)d_in[5];
    const float* pool_norm_w  = (const float*)d_in[6];
    const float* x_norm_w     = (const float*)d_in[7];
    const float* pool_qkv_w   = (const float*)d_in[8];
    const float* pool_qkv_b   = (const float*)d_in[9];
    const float* x_qkv_w      = (const float*)d_in[10];
    const float* x_qkv_b      = (const float*)d_in[11];
    const float* pool_q_norm_w = (const float*)d_in[12];
    const float* pool_k_norm_w = (const float*)d_in[13];
    const float* x_q_norm_w    = (const float*)d_in[14];
    const float* x_k_norm_w    = (const float*)d_in[15];
    const float* pool_out_w   = (const float*)d_in[16];
    const float* pool_out_b   = (const float*)d_in[17];
    const float* x_out_w      = (const float*)d_in[18];
    const float* x_out_b      = (const float*)d_in[19];
    float* out = (float*)d_out;

    // workspace layout (ushort elements)
    unsigned short* base = (unsigned short*)d_ws;
    size_t off = 0;
    unsigned short* WqkvP = base + off; off += (size_t)C3 * DIMX;     // 3072x1024
    unsigned short* WqkvX = base + off; off += (size_t)C3 * DIMX;
    unsigned short* WoutP = base + off; off += (size_t)DIMX * DIMX;   // 1024x1024
    unsigned short* WoutX = base + off; off += (size_t)DIMX * DIMX;
    unsigned short* Apool = base + off; off += (size_t)BB * PP * DIMX;
    unsigned short* Ax    = base + off; off += (size_t)BB * TTOK * DIMX;
    unsigned short* Qb    = base + off; off += (size_t)BB * NH * SSEQ * HDIM;
    unsigned short* Kb    = base + off; off += (size_t)BB * NH * SSEQ * HDIM;
    unsigned short* Vt    = base + off; off += (size_t)BB * NH * HDIM * SSEQ;
    unsigned short* attnA = base + off; off += (size_t)BB * SSEQ * DIMX;

    // 1) weights -> bf16, transposed (N-major)
    k_transpose_bf16<<<(DIMX * C3 + 255) / 256, 256, 0, stream>>>(pool_qkv_w, WqkvP, DIMX, C3);
    k_transpose_bf16<<<(DIMX * C3 + 255) / 256, 256, 0, stream>>>(x_qkv_w,    WqkvX, DIMX, C3);
    k_transpose_bf16<<<(DIMX * DIMX + 255) / 256, 256, 0, stream>>>(pool_out_w, WoutP, DIMX, DIMX);
    k_transpose_bf16<<<(DIMX * DIMX + 255) / 256, 256, 0, stream>>>(x_out_w,    WoutX, DIMX, DIMX);

    // 2) RMSNorm -> bf16 activations
    k_rmsnorm_bf16<<<BB * PP,   256, 0, stream>>>(pool, pool_norm_w, Apool);
    k_rmsnorm_bf16<<<BB * TTOK, 256, 0, stream>>>(x,    x_norm_w,    Ax);

    // 3) QKV projection + head-norm + RoPE (fused epilogue)
    {
        int nwP = (BB * PP / 16) * 48;      // 768
        int nwX = (BB * TTOK / 16) * 48;    // 12288
        k_qkv_wmma<<<(nwP + 7) / 8, 256, 0, stream>>>(
            Apool, WqkvP, pool_qkv_b, pool_q_norm_w, pool_k_norm_w,
            Qb, Kb, Vt, PP, 0, nwP);
        k_qkv_wmma<<<(nwX + 7) / 8, 256, 0, stream>>>(
            Ax, WqkvX, x_qkv_b, x_q_norm_w, x_k_norm_w,
            Qb, Kb, Vt, TTOK, PP, nwX);
    }

    // 4) attention (flash-style, online softmax)
    {
        int nw = BB * NH * (SSEQ / 16);     // 4352
        k_attn_wmma<<<(nw + 7) / 8, 256, 0, stream>>>(Qb, Kb, Vt, attn_mask, attnA, nw);
    }

    // 5) output projections + gating
    {
        int nwP = (BB * PP / 16) * 16;      // 256
        int nwX = (BB * TTOK / 16) * 16;    // 4096
        k_outproj_wmma<<<(nwP + 7) / 8, 256, 0, stream>>>(
            attnA, WoutP, pool_out_b, n_mask, out, PP, 0, nwP);
        k_outproj_wmma<<<(nwX + 7) / 8, 256, 0, stream>>>(
            attnA, WoutX, x_out_b, t_mask, out + (size_t)BB * PP * DIMX, TTOK, PP, nwX);
    }
}
